// MultiHeadAttention_7103875908186
// MI455X (gfx1250) — compile-verified, matrix operand order checked by reading, not run on hardware
//
#include <hip/hip_runtime.h>
#include <hip/hip_bf16.h>

// ---------------------------------------------------------------------------
// Shapes (compile-time constants from the reference)
// ---------------------------------------------------------------------------
#define B_   2
#define L_   2048
#define D_   1536
#define H_   24
#define DH_  64
#define M_TOK (B_ * L_)          // 4096 tokens
#define QKV_N (3 * D_)           // 4608

typedef __attribute__((ext_vector_type(16))) __bf16 v16bf;
typedef __attribute__((ext_vector_type(8)))  float  v8f;
typedef __attribute__((ext_vector_type(4)))  unsigned int v4u;
typedef __attribute__((ext_vector_type(8)))  int    v8i;
typedef __attribute__((ext_vector_type(4)))  int    v4i;

union F16x16 { v16bf v; uint4 q[2]; __bf16 e[16]; };
union F8f    { v8f   v; float e[8]; };

#if defined(__has_builtin)
#if __has_builtin(__builtin_amdgcn_tensor_load_to_lds) && \
    __has_builtin(__builtin_amdgcn_s_wait_tensorcnt)
#define USE_TDM 1
#endif
#endif

__device__ __forceinline__ v8f wmma_bf16(v16bf a, v16bf b, v8f c) {
  // (neg_a, A, neg_b, B, c_mod, C, reuse_a, reuse_b)
  return __builtin_amdgcn_wmma_f32_16x16x32_bf16(false, a, false, b,
                                                 (short)0, c, false, false);
}

__device__ __forceinline__ v8f zero8() {
  F8f z;
#pragma unroll
  for (int i = 0; i < 8; ++i) z.e[i] = 0.0f;
  return z.v;
}

// Load a 16x32 bf16 fragment (A-layout; B uses the same pattern with rows = N,
// contiguous along K).  base points at tile origin, ld = row stride in elems.
// Lane l: row = l&15, K-chunks {kb..kb+7} and {kb+16..kb+23}, kb = 8*(l>=16).
__device__ __forceinline__ v16bf ld_frag(const __bf16* base, int ld) {
  int lane = threadIdx.x & 31;
  int r  = lane & 15;
  int kb = (lane >> 4) << 3;
  F16x16 f;
  f.q[0] = *(const uint4*)(base + (size_t)r * ld + kb);
  f.q[1] = *(const uint4*)(base + (size_t)r * ld + kb + 16);
  return f.v;
}

// ---------------------------------------------------------------------------
// Kernel 1: h = LayerNorm(x) * w + b, output bf16.  One block per token.
// ---------------------------------------------------------------------------
__global__ __launch_bounds__(256)
void ln_to_bf16(const float* __restrict__ x, const float* __restrict__ w,
                const float* __restrict__ b, __bf16* __restrict__ out) {
  int t = blockIdx.x;
  const float* xr = x + (size_t)t * D_;
  float v[6], s = 0.f, ss = 0.f;
#pragma unroll
  for (int i = 0; i < 6; ++i) {
    v[i] = xr[threadIdx.x + 256 * i];
    s += v[i]; ss += v[i] * v[i];
  }
#pragma unroll
  for (int m = 16; m > 0; m >>= 1) {
    s  += __shfl_xor(s, m, 32);
    ss += __shfl_xor(ss, m, 32);
  }
  __shared__ float ps[8], pss[8];
  if ((threadIdx.x & 31) == 0) { ps[threadIdx.x >> 5] = s; pss[threadIdx.x >> 5] = ss; }
  __syncthreads();
  s = 0.f; ss = 0.f;
#pragma unroll
  for (int i = 0; i < 8; ++i) { s += ps[i]; ss += pss[i]; }
  float mean = s * (1.0f / D_);
  float var  = ss * (1.0f / D_) - mean * mean;
  float rstd = rsqrtf(var + 1e-5f);
#pragma unroll
  for (int i = 0; i < 6; ++i) {
    int d = threadIdx.x + 256 * i;
    out[(size_t)t * D_ + d] = (__bf16)((v[i] - mean) * rstd * w[d] + b[d]);
  }
}

// ---------------------------------------------------------------------------
// Kernel 2/5: C[M,N] = A[M,K](bf16, row-major) * W[N,K](f32, row-major)^T
// 128x128 block tile, BK=32, 8 waves (2x4), wave tile 64x32 (8 WMMA accums).
// Double-buffered LDS with register-staged prefetch; one barrier per k-step.
// N, K compile-time for cheap address math.
// ---------------------------------------------------------------------------
template<int N, int K>
__global__ __launch_bounds__(256)
void gemm_bf16f32(const __bf16* __restrict__ A, const float* __restrict__ W,
                  float* __restrict__ C) {
  __shared__ __align__(16) __bf16 As[2][128][32];
  __shared__ __align__(16) __bf16 Bs[2][128][32];
  int tid = threadIdx.x, wid = tid >> 5, lane = tid & 31;
  int hl = lane >> 4, ln = lane & 15;
  int bm = blockIdx.y * 128, bn = blockIdx.x * 128;
  int wm = (wid >> 2) * 64;      // 0 / 64
  int wn = (wid & 3) * 32;       // 0,32,64,96

  // per-thread tile-load coordinates (A: 2 x b128 chunks, W: 4 x float4)
  int arow[2], acol[2], wrow[4], wcol[4];
#pragma unroll
  for (int j = 0; j < 2; ++j) {
    int c = tid * 2 + j;
    arow[j] = c >> 2; acol[j] = (c & 3) * 8;
  }
#pragma unroll
  for (int j = 0; j < 4; ++j) {
    int c = tid * 4 + j;
    wrow[j] = c >> 3; wcol[j] = (c & 7) * 4;
  }

  auto loadA = [&](int k0, uint4* ar) {
#pragma unroll
    for (int j = 0; j < 2; ++j)
      ar[j] = *(const uint4*)(A + (size_t)(bm + arow[j]) * K + k0 + acol[j]);
  };
  auto loadW = [&](int k0, float4* wr) {
#pragma unroll
    for (int j = 0; j < 4; ++j)
      wr[j] = *(const float4*)(W + (size_t)(bn + wrow[j]) * K + k0 + wcol[j]);
  };
  auto stage = [&](int buf, const uint4* ar, const float4* wr) {
#pragma unroll
    for (int j = 0; j < 2; ++j)
      *(uint4*)&As[buf][arow[j]][acol[j]] = ar[j];
#pragma unroll
    for (int j = 0; j < 4; ++j) {
      Bs[buf][wrow[j]][wcol[j] + 0] = (__bf16)wr[j].x;
      Bs[buf][wrow[j]][wcol[j] + 1] = (__bf16)wr[j].y;
      Bs[buf][wrow[j]][wcol[j] + 2] = (__bf16)wr[j].z;
      Bs[buf][wrow[j]][wcol[j] + 3] = (__bf16)wr[j].w;
    }
  };

  v8f acc[4][2];
#pragma unroll
  for (int i = 0; i < 4; ++i)
#pragma unroll
    for (int j = 0; j < 2; ++j) acc[i][j] = zero8();

  uint4  ar[2];
  float4 wr[4];
  loadA(0, ar); loadW(0, wr);
  stage(0, ar, wr);
  __syncthreads();

  for (int k0 = 0; k0 < K; k0 += 32) {
    int cur = (k0 >> 5) & 1;
    bool more = (k0 + 32) < K;
    if (more) { loadA(k0 + 32, ar); loadW(k0 + 32, wr); }

    v16bf bf0 = ld_frag(&Bs[cur][wn][0], 32);
    v16bf bf1 = ld_frag(&Bs[cur][wn + 16][0], 32);
#pragma unroll
    for (int i = 0; i < 4; ++i) {
      v16bf a = ld_frag(&As[cur][wm + 16 * i][0], 32);
      acc[i][0] = wmma_bf16(a, bf0, acc[i][0]);
      acc[i][1] = wmma_bf16(a, bf1, acc[i][1]);
    }
    if (more) stage(cur ^ 1, ar, wr);
    __syncthreads();
  }

#pragma unroll
  for (int i = 0; i < 4; ++i)
#pragma unroll
    for (int j = 0; j < 2; ++j) {
      F8f f; f.v = acc[i][j];
#pragma unroll
      for (int e = 0; e < 8; ++e) {
        int r = bm + wm + 16 * i + e + 8 * hl;
        int c = bn + wn + 16 * j + ln;
        C[(size_t)r * N + c] = f.e[e];
      }
    }
}

// ---------------------------------------------------------------------------
// Kernel 3: q/k LayerNorm (over D) + RoPE + pack q,k,v -> [B,H,L,DH] bf16.
// One block per token.
// ---------------------------------------------------------------------------
__global__ __launch_bounds__(256)
void qk_rope_pack(const float* __restrict__ qkv,
                  const float* __restrict__ qw, const float* __restrict__ kw,
                  __bf16* __restrict__ Qo, __bf16* __restrict__ Ko,
                  __bf16* __restrict__ Vo) {
  int t = blockIdx.x;
  int bb = t / L_, l = t % L_;
  const float* qr = qkv + (size_t)t * QKV_N;
  const float* kr = qr + D_;
  const float* vr = qr + 2 * D_;
  __shared__ float qs[D_], ks[D_];
  float sq = 0.f, sqq = 0.f, sk = 0.f, skk = 0.f;
#pragma unroll
  for (int i = 0; i < 6; ++i) {
    int d = threadIdx.x + 256 * i;
    float a = qr[d], c = kr[d];
    qs[d] = a; ks[d] = c;
    sq += a; sqq += a * a; sk += c; skk += c * c;
  }
#pragma unroll
  for (int m = 16; m > 0; m >>= 1) {
    sq += __shfl_xor(sq, m, 32);   sqq += __shfl_xor(sqq, m, 32);
    sk += __shfl_xor(sk, m, 32);   skk += __shfl_xor(skk, m, 32);
  }
  __shared__ float r0[8], r1[8], r2[8], r3[8];
  if ((threadIdx.x & 31) == 0) {
    int w = threadIdx.x >> 5;
    r0[w] = sq; r1[w] = sqq; r2[w] = sk; r3[w] = skk;
  }
  __syncthreads();                      // also publishes qs[], ks[]
  sq = sqq = sk = skk = 0.f;
#pragma unroll
  for (int i = 0; i < 8; ++i) { sq += r0[i]; sqq += r1[i]; sk += r2[i]; skk += r3[i]; }
  float qm = sq * (1.0f / D_), km = sk * (1.0f / D_);
  float qrstd = rsqrtf(sqq * (1.0f / D_) - qm * qm + 1e-5f);
  float krstd = rsqrtf(skk * (1.0f / D_) - km * km + 1e-5f);

  const float kfreq = 0.28782313662425572f;  // ln(10000)/32
#pragma unroll
  for (int i = 0; i < 6; ++i) {
    int d  = threadIdx.x + 256 * i;
    int h  = d >> 6, dh = d & 63;
    int fi = dh & 31;
    float ang = (float)l * __expf(-(float)fi * kfreq);
    float co = __cosf(ang), si = __sinf(ang);
    int dp = (dh < 32) ? d + 32 : d - 32;
    float qn = (qs[d]  - qm) * qrstd * qw[d];
    float qp = (qs[dp] - qm) * qrstd * qw[dp];
    float kn = (ks[d]  - km) * krstd * kw[d];
    float kp = (ks[dp] - km) * krstd * kw[dp];
    float qo = (dh < 32) ? (qn * co - qp * si) : (qn * co + qp * si);
    float ko = (dh < 32) ? (kn * co - kp * si) : (kn * co + kp * si);
    size_t o = (((size_t)(bb * H_ + h)) * L_ + l) * DH_ + dh;
    Qo[o] = (__bf16)qo;
    Ko[o] = (__bf16)ko;
    Vo[o] = (__bf16)vr[d];
  }
}

// ---------------------------------------------------------------------------
// Kernel 4: flash attention with seq_id block-diagonal mask.
// Block: 128 queries of one (b,h); 8 waves x 16 query rows; 64-key tiles.
// K tile comes in via the Tensor Data Mover (wave 0 issues the descriptor,
// waits on TENSORcnt) while the other waves transpose V into LDS.
// ---------------------------------------------------------------------------
__global__ __launch_bounds__(256)
void flash_attn(const __bf16* __restrict__ Qg, const __bf16* __restrict__ Kg,
                const __bf16* __restrict__ Vg, const int* __restrict__ seq_id,
                __bf16* __restrict__ ctx) {
  int q0 = blockIdx.x * 128;
  int h  = blockIdx.y;
  int bb = blockIdx.z;
  size_t bh = ((size_t)bb * H_ + h) * L_;
  int tid = threadIdx.x, wid = tid >> 5, lane = tid & 31;
  int hl = lane >> 4, ln = lane & 15;

  __shared__ __align__(16) __bf16 Ks[64][64];        // keys row-major
  __shared__ __align__(16) __bf16 Vt[64][64];        // V transposed: [dh][key]
  __shared__ __align__(16) __bf16 Ps[8][16][64];     // per-wave P scratch
  __shared__ int sq[128];
  __shared__ int sk[64];

  // Q fragments for this wave's 16 query rows (two K=32 chunks of DH=64)
  int qw0 = q0 + wid * 16;
  v16bf aq0 = ld_frag(Qg + (bh + qw0) * DH_, DH_);
  v16bf aq1 = ld_frag(Qg + (bh + qw0) * DH_ + 32, DH_);

  if (tid < 128) sq[tid] = seq_id[(size_t)bb * L_ + q0 + tid];

  float mi[8], li[8];
#pragma unroll
  for (int e = 0; e < 8; ++e) { mi[e] = -3e38f; li[e] = 0.f; }
  v8f o[4];
#pragma unroll
  for (int j = 0; j < 4; ++j) o[j] = zero8();

  for (int k0 = 0; k0 < L_; k0 += 64) {
    __syncthreads();
#if defined(USE_TDM)
    if (wid == 0) {
      // Tensor DMA: 2-D tile 64x64 of bf16 from K[bh + k0 ..] into Ks.
      unsigned lds_off = (unsigned)(unsigned long long)(const void*)&Ks[0][0];
      unsigned long long ga =
          (unsigned long long)(const void*)(Kg + (bh + k0) * DH_);
      v4u g0;
      g0.x = 1u;                                   // count=1, user descriptor
      g0.y = lds_off;                              // lds_addr
      g0.z = (unsigned)ga;                         // global_addr[31:0]
      g0.w = (unsigned)((ga >> 32) & 0x1FFFFFFu) | (2u << 30);  // addr hi, type=2
      v8i g1;
      g1[0] = 0x00010000;          // wg_mask=0, data_size=1 (2 bytes)
      g1[1] = (int)(64u << 16);    // tensor_dim0 = 64 (bits 79:48)
      g1[2] = (int)(2048u << 16);  // tensor_dim1 = 2048 (bits 111:80)
      g1[3] = (int)(64u << 16);    // tile_dim0 = 64 (bits 127:112)
      g1[4] = 64;                  // tile_dim1 = 64
      g1[5] = 64;                  // tensor_dim0_stride = 64
      g1[6] = 0; g1[7] = 0;
      v4i zz; zz[0] = 0; zz[1] = 0; zz[2] = 0; zz[3] = 0;
#if __clang_major__ >= 23
      v8i z8; z8[0]=0; z8[1]=0; z8[2]=0; z8[3]=0; z8[4]=0; z8[5]=0; z8[6]=0; z8[7]=0;
      __builtin_amdgcn_tensor_load_to_lds(g0, g1, zz, zz, z8, 0);
#else
      __builtin_amdgcn_tensor_load_to_lds(g0, g1, zz, zz, 0);
#endif
      __builtin_amdgcn_s_wait_tensorcnt(0);
    }
#else
#pragma unroll
    for (int j = 0; j < 2; ++j) {            // K tile 64x64 bf16
      int c = tid * 2 + j;
      int r = c >> 3, cc = (c & 7) * 8;
      *(uint4*)&Ks[r][cc] = *(const uint4*)(Kg + (bh + k0 + r) * DH_ + cc);
    }
#endif
#pragma unroll
    for (int j = 0; j < 2; ++j) {            // V tile, transposed into LDS
      int c = tid * 2 + j;
      int r = c >> 3, cc = (c & 7) * 8;
      union { uint4 u; __bf16 e[8]; } uu;
      uu.u = *(const uint4*)(Vg + (bh + k0 + r) * DH_ + cc);
#pragma unroll
      for (int x = 0; x < 8; ++x) Vt[cc + x][r] = uu.e[x];
    }
    if (tid < 64) sk[tid] = seq_id[(size_t)bb * L_ + k0 + tid];
    __syncthreads();

    // S = (Q Kt) * scale, masked
    float p[4][8], rowmax[8];
#pragma unroll
    for (int e = 0; e < 8; ++e) rowmax[e] = -3e38f;
#pragma unroll
    for (int j = 0; j < 4; ++j) {
      v8f s = zero8();
      v16bf bk0 = ld_frag(&Ks[16 * j][0], 64);
      v16bf bk1 = ld_frag(&Ks[16 * j][32], 64);
      s = wmma_bf16(aq0, bk0, s);
      s = wmma_bf16(aq1, bk1, s);
      F8f f; f.v = s;
      int sid_k = sk[16 * j + ln];
#pragma unroll
      for (int e = 0; e < 8; ++e) {
        int mrow = wid * 16 + e + 8 * hl;
        bool valid = (sq[mrow] == sid_k);
        float sv = valid ? f.e[e] * 0.125f : -1e30f;   // 1/sqrt(64)
        p[j][e] = sv;
        rowmax[e] = fmaxf(rowmax[e], sv);
      }
    }
#pragma unroll
    for (int m = 1; m < 16; m <<= 1)
#pragma unroll
      for (int e = 0; e < 8; ++e)
        rowmax[e] = fmaxf(rowmax[e], __shfl_xor(rowmax[e], m, 32));

    float alpha[8], rs[8];
#pragma unroll
    for (int e = 0; e < 8; ++e) {
      float mn = fmaxf(mi[e], rowmax[e]);
      alpha[e] = __expf(mi[e] - mn);
      mi[e] = mn;
      rs[e] = 0.f;
    }
#pragma unroll
    for (int j = 0; j < 4; ++j)
#pragma unroll
      for (int e = 0; e < 8; ++e) {
        float pe = (p[j][e] <= -1e29f) ? 0.f : __expf(p[j][e] - mi[e]);
        p[j][e] = pe;
        rs[e] += pe;
      }
#pragma unroll
    for (int m = 1; m < 16; m <<= 1)
#pragma unroll
      for (int e = 0; e < 8; ++e) rs[e] += __shfl_xor(rs[e], m, 32);
#pragma unroll
    for (int e = 0; e < 8; ++e) li[e] = li[e] * alpha[e] + rs[e];
#pragma unroll
    for (int j = 0; j < 4; ++j) {
      F8f f; f.v = o[j];
#pragma unroll
      for (int e = 0; e < 8; ++e) f.e[e] *= alpha[e];
      o[j] = f.v;
    }
    // P -> per-wave LDS scratch in A-fragment-readable layout
#pragma unroll
    for (int j = 0; j < 4; ++j)
#pragma unroll
      for (int e = 0; e < 8; ++e)
        Ps[wid][e + 8 * hl][16 * j + ln] = (__bf16)p[j][e];

    v16bf ap0 = ld_frag(&Ps[wid][0][0], 64);
    v16bf ap1 = ld_frag(&Ps[wid][0][32], 64);
#pragma unroll
    for (int jd = 0; jd < 4; ++jd) {
      v16bf bv0 = ld_frag(&Vt[16 * jd][0], 64);
      v16bf bv1 = ld_frag(&Vt[16 * jd][32], 64);
      o[jd] = wmma_bf16(ap0, bv0, o[jd]);
      o[jd] = wmma_bf16(ap1, bv1, o[jd]);
    }
  }
  // epilogue: ctx[b, l, h*64 + dh] bf16
#pragma unroll
  for (int jd = 0; jd < 4; ++jd) {
    F8f f; f.v = o[jd];
#pragma unroll
    for (int e = 0; e < 8; ++e) {
      int row = q0 + wid * 16 + e + 8 * hl;
      int col = h * DH_ + 16 * jd + ln;
      ctx[((size_t)bb * L_ + row) * D_ + col] = (__bf16)(f.e[e] / li[e]);
    }
  }
}

// ---------------------------------------------------------------------------
// Host launcher
// ---------------------------------------------------------------------------
extern "C" void kernel_launch(void* const* d_in, const int* in_sizes, int n_in,
                              void* d_out, int out_size, void* d_ws, size_t ws_size,
                              hipStream_t stream) {
  const float* x     = (const float*)d_in[0];
  const int*   sid   = (const int*)  d_in[1];
  const float* ln_w  = (const float*)d_in[2];
  const float* ln_b  = (const float*)d_in[3];
  const float* w_qkv = (const float*)d_in[4];
  const float* q_ln  = (const float*)d_in[5];
  const float* k_ln  = (const float*)d_in[6];
  const float* w_out = (const float*)d_in[7];
  float* out = (float*)d_out;

  char* p = (char*)d_ws;
  const size_t SZ_BF = (size_t)M_TOK * D_ * sizeof(__bf16);   // 12.6 MB
  __bf16* h   = (__bf16*)p;  p += SZ_BF;
  float*  qkv = (float*)p;   p += (size_t)M_TOK * QKV_N * sizeof(float);
  __bf16* Qb  = (__bf16*)p;  p += SZ_BF;
  __bf16* Kb  = (__bf16*)p;  p += SZ_BF;
  __bf16* Vb  = (__bf16*)p;  p += SZ_BF;
  __bf16* ctx = (__bf16*)p;  p += SZ_BF;

  ln_to_bf16<<<M_TOK, 256, 0, stream>>>(x, ln_w, ln_b, h);

  gemm_bf16f32<QKV_N, D_><<<dim3(QKV_N / 128, M_TOK / 128), 256, 0, stream>>>(
      h, w_qkv, qkv);

  qk_rope_pack<<<M_TOK, 256, 0, stream>>>(qkv, q_ln, k_ln, Qb, Kb, Vb);

  flash_attn<<<dim3(L_ / 128, H_, B_), 256, 0, stream>>>(Qb, Kb, Vb, sid, ctx);

  gemm_bf16f32<D_, D_><<<dim3(D_ / 128, M_TOK / 128), 256, 0, stream>>>(
      ctx, w_out, out);
}
